// LSTMLayer_36833639531104
// MI455X (gfx1250) — matrix operand order repaired and checked by reading, not compiled, not run on hardware
//
#include <hip/hip_runtime.h>
#include <hip/hip_bf16.h>
#include <math.h>

// ---------------------------------------------------------------------------
// GRU-style fused cell on MI455X (gfx1250), bf16 WMMA path.
//   GEMM1: [X|S] (B x 1280) @ [U3;W3]^T-panels -> tanh -> Z,G, SR=R*S
//   GEMM2: [X|SR] (B x 1280) @ [Uh;Wh]         -> tanh -> (1-G)*H + Z*S
// Wave tile 32x64 (8 WMMA accum tiles): ~160 VGPRs -> stays under the 256
// direct-address ceiling (no s_set_vgpr_msb / v_mov shuffle overhead) and
// leaves occupancy headroom to hide LDS latency.
// ---------------------------------------------------------------------------

typedef __attribute__((ext_vector_type(16))) __bf16 v16bf;
typedef __attribute__((ext_vector_type(8)))  float  v8f;
typedef __attribute__((ext_vector_type(4)))  float  vf4;   // trivial 16B vector

#define B_ROWS 32768
#define D_IN   256
#define D_OUT  1024
#define KTOT   1280      // D_IN + D_OUT
#define N1     3072      // 3*D_OUT (Z|G|R)
#define N2     1024

#define TM     128
#define TN     128
#define TK     32
#define KPAD   40        // padded LDS k-stride (80B) -> conflict-free b128 loads

union AFrag { v16bf v; vf4 q[2]; };

// ------------------------- conversion kernels ------------------------------

__global__ __launch_bounds__(256) void cvt_A(const float* __restrict__ X,
                                             const float* __restrict__ S,
                                             __bf16* __restrict__ A1,
                                             __bf16* __restrict__ A2) {
  size_t i = (size_t)blockIdx.x * blockDim.x + threadIdx.x;
  if (i >= (size_t)B_ROWS * KTOT) return;
  size_t b = i / KTOT;
  int    k = (int)(i % KTOT);
  float  v = (k < D_IN) ? X[b * D_IN + k] : S[b * D_OUT + (k - D_IN)];
  A1[i] = (__bf16)v;
  if (k < D_IN) A2[i] = (__bf16)v;   // SR part of A2 is filled by GEMM1 epilogue
}

__global__ __launch_bounds__(256) void cvt_B1(const float* __restrict__ Uz,
                                              const float* __restrict__ Ug,
                                              const float* __restrict__ Ur,
                                              const float* __restrict__ Wz,
                                              const float* __restrict__ Wg,
                                              const float* __restrict__ Wr,
                                              __bf16* __restrict__ B1t) {
  size_t i = (size_t)blockIdx.x * blockDim.x + threadIdx.x;
  if (i >= (size_t)N1 * KTOT) return;
  int n = (int)(i / KTOT);
  int k = (int)(i % KTOT);
  int seg = n >> 10, nn = n & 1023;
  const float* U = (seg == 0) ? Uz : (seg == 1) ? Ug : Ur;
  const float* W = (seg == 0) ? Wz : (seg == 1) ? Wg : Wr;
  float v = (k < D_IN) ? U[(size_t)k * D_OUT + nn]
                       : W[(size_t)(k - D_IN) * D_OUT + nn];
  B1t[i] = (__bf16)v;   // stored "transposed": row n holds K-contiguous data
}

__global__ __launch_bounds__(256) void cvt_B2(const float* __restrict__ Uh,
                                              const float* __restrict__ Wh,
                                              __bf16* __restrict__ B2t) {
  size_t i = (size_t)blockIdx.x * blockDim.x + threadIdx.x;
  if (i >= (size_t)N2 * KTOT) return;
  int n = (int)(i / KTOT);
  int k = (int)(i % KTOT);
  float v = (k < D_IN) ? Uh[(size_t)k * D_OUT + n]
                       : Wh[(size_t)(k - D_IN) * D_OUT + n];
  B2t[i] = (__bf16)v;
}

// --------------------------- fused bf16 GEMM -------------------------------
// MODE 1: epilogue = tanh(+b3) -> Z,G bf16 ; SR = R*S into A2[:,256+n]
// MODE 2: epilogue = tanh(+bh) -> out = (1-G)*H + Z*S (f32)

template <int MODE>
__global__ __launch_bounds__(256) void gemm_fused(
    const __bf16* __restrict__ A, const __bf16* __restrict__ Bt,
    const float* __restrict__ bz, const float* __restrict__ bg,
    const float* __restrict__ br, const float* __restrict__ bh,
    const float* __restrict__ Sf,
    __bf16* __restrict__ Z, __bf16* __restrict__ G,
    __bf16* __restrict__ A2sr, float* __restrict__ out) {
  __shared__ __bf16 As[2][TM * KPAD];   // 2 x 10 KB
  __shared__ __bf16 Bs[2][TN * KPAD];   // 2 x 10 KB

  const int tid  = threadIdx.x;
  const int lane = tid & 31;
  const int wave = tid >> 5;
  const int wm   = wave >> 1;        // 0..3 : wave M offset = wm*32
  const int wn   = wave & 1;         // 0..1 : wave N offset = wn*64
  const int rsel = lane & 15;
  const int hi   = lane >> 4;        // ISA half-wave select

  const int blockM = blockIdx.y * TM;
  const int blockN = blockIdx.x * TN;

  // cooperative tile-load mapping: 64 rows x 4 k-groups of 8 bf16 (16B)
  const int lr = tid >> 2;           // 0..63
  const int lk = (tid & 3) * 8;      // 0,8,16,24

  const __bf16* gA = A  + (size_t)(blockM + lr) * KTOT + lk;
  const __bf16* gB = Bt + (size_t)(blockN + lr) * KTOT + lk;

  vf4 ra0, ra1, rb0, rb1;
#define LOAD_REGS(kt)                                                   \
  do {                                                                  \
    size_t ko = (size_t)(kt)*TK;                                        \
    ra0 = *(const vf4*)(gA + ko);                                       \
    ra1 = *(const vf4*)(gA + (size_t)64 * KTOT + ko);                   \
    rb0 = *(const vf4*)(gB + ko);                                       \
    rb1 = *(const vf4*)(gB + (size_t)64 * KTOT + ko);                   \
  } while (0)
#define STORE_LDS(buf)                                                  \
  do {                                                                  \
    *(vf4*)&As[buf][lr * KPAD + lk]        = ra0;                       \
    *(vf4*)&As[buf][(lr + 64) * KPAD + lk] = ra1;                       \
    *(vf4*)&Bs[buf][lr * KPAD + lk]        = rb0;                       \
    *(vf4*)&Bs[buf][(lr + 64) * KPAD + lk] = rb1;                       \
  } while (0)

  v8f acc[2][4] = {};
  const int nk = KTOT / TK;   // 40

  LOAD_REGS(0);
  STORE_LDS(0);
  __syncthreads();

  for (int kt = 0; kt < nk; ++kt) {
    const int c = kt & 1;
    if (kt + 1 < nk) LOAD_REGS(kt + 1);

    // A fragment (16x32 bf16): lane<16 -> K{0..7,16..23}; lane>=16 -> K{8..15,24..31}
    AFrag af[2];
#pragma unroll
    for (int t = 0; t < 2; ++t) {
      int row = wm * 32 + t * 16 + rsel;
      af[t].q[0] = *(const vf4*)&As[c][row * KPAD + hi * 8];
      af[t].q[1] = *(const vf4*)&As[c][row * KPAD + hi * 8 + 16];
    }
    // B fragment (32x16 bf16) from transposed panel: lane<16 -> K{0..7,16..23}; lane>=16 -> K{8..15,24..31}
    AFrag bfr[4];
#pragma unroll
    for (int u = 0; u < 4; ++u) {
      int col = wn * 64 + u * 16 + rsel;
      bfr[u].q[0] = *(const vf4*)&Bs[c][col * KPAD + hi * 8];
      bfr[u].q[1] = *(const vf4*)&Bs[c][col * KPAD + hi * 8 + 16];
    }

#pragma unroll
    for (int t = 0; t < 2; ++t)
#pragma unroll
      for (int u = 0; u < 4; ++u)
        acc[t][u] = __builtin_amdgcn_wmma_f32_16x16x32_bf16(
            false, af[t].v, false, bfr[u].v, (short)0, acc[t][u], false, false);

    if (kt + 1 < nk) {
      STORE_LDS((kt + 1) & 1);
      __syncthreads();
    }
  }

  // ----------------------------- epilogue ---------------------------------
#pragma unroll
  for (int t = 0; t < 2; ++t) {
#pragma unroll
    for (int u = 0; u < 4; ++u) {
#pragma unroll
      for (int i = 0; i < 8; ++i) {
        int m = blockM + wm * 32 + t * 16 + hi * 8 + i;   // C layout: M = i + 8*hi
        int n = blockN + wn * 64 + u * 16 + rsel;         // N = lane&15
        float v = acc[t][u][i];
        if (MODE == 1) {
          int seg = n >> 10;        // uniform per block (TN divides 1024)
          int nn  = n & 1023;
          float bias = (seg == 0) ? bz[nn] : (seg == 1) ? bg[nn] : br[nn];
          float a = tanhf(v + bias);
          size_t idx = (size_t)m * D_OUT + nn;
          if (seg == 0)      Z[idx] = (__bf16)a;
          else if (seg == 1) G[idx] = (__bf16)a;
          else               A2sr[(size_t)m * KTOT + D_IN + nn] =
                                 (__bf16)(a * Sf[idx]);   // SR = R * S
        } else {
          float h = tanhf(v + bh[n]);
          size_t idx = (size_t)m * D_OUT + n;
          float g = (float)G[idx];
          float z = (float)Z[idx];
          float s = Sf[idx];
          out[idx] = (1.0f - g) * h + z * s;
        }
      }
    }
  }
#undef LOAD_REGS
#undef STORE_LDS
}

// ------------------------------ launcher -----------------------------------

extern "C" void kernel_launch(void* const* d_in, const int* in_sizes, int n_in,
                              void* d_out, int out_size, void* d_ws, size_t ws_size,
                              hipStream_t stream) {
  const float* S  = (const float*)d_in[0];
  const float* X  = (const float*)d_in[1];
  const float* Uz = (const float*)d_in[2];
  const float* Ug = (const float*)d_in[3];
  const float* Ur = (const float*)d_in[4];
  const float* Uh = (const float*)d_in[5];
  const float* Wz = (const float*)d_in[6];
  const float* Wg = (const float*)d_in[7];
  const float* Wr = (const float*)d_in[8];
  const float* Wh = (const float*)d_in[9];
  const float* bz = (const float*)d_in[10];
  const float* bg = (const float*)d_in[11];
  const float* br = (const float*)d_in[12];
  const float* bh = (const float*)d_in[13];
  float* out = (float*)d_out;

  // workspace layout (bf16), ~312 MB total
  __bf16* ws  = (__bf16*)d_ws;
  __bf16* A1  = ws;                                   // B x 1280
  __bf16* A2  = A1  + (size_t)B_ROWS * KTOT;          // B x 1280
  __bf16* B1t = A2  + (size_t)B_ROWS * KTOT;          // 3072 x 1280 (transposed)
  __bf16* B2t = B1t + (size_t)N1 * KTOT;              // 1024 x 1280 (transposed)
  __bf16* Zb  = B2t + (size_t)N2 * KTOT;              // B x 1024
  __bf16* Gb  = Zb  + (size_t)B_ROWS * D_OUT;         // B x 1024

  {
    size_t n = (size_t)B_ROWS * KTOT;
    cvt_A<<<(unsigned)((n + 255) / 256), 256, 0, stream>>>(X, S, A1, A2);
  }
  {
    size_t n = (size_t)N1 * KTOT;
    cvt_B1<<<(unsigned)((n + 255) / 256), 256, 0, stream>>>(Uz, Ug, Ur, Wz, Wg, Wr, B1t);
  }
  {
    size_t n = (size_t)N2 * KTOT;
    cvt_B2<<<(unsigned)((n + 255) / 256), 256, 0, stream>>>(Uh, Wh, B2t);
  }

  dim3 g1(N1 / TN, B_ROWS / TM);   // 24 x 256
  gemm_fused<1><<<g1, 256, 0, stream>>>(A1, B1t, bz, bg, br, bh, S,
                                        Zb, Gb, A2, nullptr);

  dim3 g2(N2 / TN, B_ROWS / TM);   // 8 x 256
  gemm_fused<2><<<g2, 256, 0, stream>>>(A2, B2t, bz, bg, br, bh, S,
                                        Zb, Gb, nullptr, out);
}
